// SparseyLayer_85186381349359
// MI455X (gfx1250) — matrix operand order repaired and checked by reading, not compiled, not run on hardware
//
#include <hip/hip_runtime.h>

typedef __attribute__((ext_vector_type(16))) _Float16 v16h;
typedef __attribute__((ext_vector_type(8)))  _Float16 v8h;
typedef __attribute__((ext_vector_type(8)))  float    v8f;

#define B_     128
#define PREVM  100
#define M_     100
#define F_     25
#define C_     8
#define N_     16
#define D_     3200          // F_ * 8 * 16
#define EPS_   1e-4f

// ---------------------------------------------------------------------------
// Kernel 1: gather + L1-normalize -> xn f16, layout (m, b, d) row-major.
// One wave (32 lanes) per (m,b) pair; 4 waves per 128-thread block.
// ---------------------------------------------------------------------------
__global__ void k_gather_norm(const float* __restrict__ x,
                              const int* __restrict__ filt,
                              _Float16* __restrict__ xn) {
  int wid  = blockIdx.x * 4 + (threadIdx.x >> 5);   // 0 .. M_*B_-1
  int lane = threadIdx.x & 31;
  int m = wid / B_;
  int b = wid - m * B_;
  const int* fm = filt + m * F_;

  // pass 1: sum of gathered 3200 elements
  float s = 0.f;
  for (int i = 0; i < D_ / 32; ++i) {
    int d = i * 32 + lane;
    int f = d >> 7;          // 128 elements per filter group
    int r = d & 127;
    int pm = fm[f];
    s += x[(size_t)(b * PREVM + pm) * 128 + r];
  }
  for (int off = 16; off >= 1; off >>= 1)
    s += __shfl_xor(s, off, 32);
  float inv = (s > 0.f) ? (1.f / s) : 0.f;

  // pass 2: normalize + convert to f16 (reads hit L0/L2)
  _Float16* out = xn + (size_t)(m * B_ + b) * D_;
  for (int i = 0; i < D_ / 32; ++i) {
    int d = i * 32 + lane;
    int f = d >> 7;
    int r = d & 127;
    float v = x[(size_t)(b * PREVM + fm[f]) * 128 + r] * inv;
    out[d] = (_Float16)v;
  }
}

// ---------------------------------------------------------------------------
// Kernel 2: weights (m,c,d,n) fp32 -> wt (m,c,n,d) f16  (N-major columns so
// WMMA B-fragments are contiguous per lane). Block: (mc, d-chunk of 128).
// ---------------------------------------------------------------------------
__global__ void k_transpose_w(const float* __restrict__ w,
                              _Float16* __restrict__ wt) {
  int d0 = blockIdx.x * 128;
  int mc = blockIdx.y;                 // 0 .. M_*C_-1
  int t  = threadIdx.x;
  // tile: 16 n x 128 d = 2048 elements, 8 per thread (write-contiguous)
  for (int j = 0; j < 8; ++j) {
    int e  = j * 256 + t;
    int n  = e >> 7;
    int dd = e & 127;
    float v = w[(size_t)(mc * D_ + d0 + dd) * N_ + n];
    wt[(size_t)(mc * N_ + n) * D_ + d0 + dd] = (_Float16)v;
  }
}

// ---------------------------------------------------------------------------
// Kernel 3: per-m GEMM  logits[m] = xn_m (128 x 3200) * W_m (3200 x 128)
// block (bt, m): 256 threads = 8 waves, wave c computes 16x16 tile
// (rows b = bt*16..+15, cols n = 0..15 of channel c) with
// v_wmma_f32_16x16x32_f16 over K = 3200.
// ---------------------------------------------------------------------------
__global__ void k_gemm(const _Float16* __restrict__ xn,
                       const _Float16* __restrict__ wt,
                       float* __restrict__ logits) {
  int bt   = blockIdx.x;               // 0..7  (b tile)
  int m    = blockIdx.y;               // 0..99
  int c    = threadIdx.x >> 5;         // wave id = channel
  int lane = threadIdx.x & 31;
  int half = lane >> 4;                // 0 or 1
  int lr   = lane & 15;

  // A: lane holds row (bt*16 + lr); K halves: half*16 + [0..16)
  const _Float16* arow = xn + (size_t)(m * B_ + bt * 16 + lr) * D_;
  int ao = half * 16;
  // B: lane holds column n = lr of channel c; K halves: half*16 + [0..16)
  const _Float16* bcol = wt + (size_t)((m * C_ + c) * N_ + lr) * D_ + half * 16;

  v8f acc{};
#pragma unroll 4
  for (int k = 0; k < D_; k += 32) {
    __builtin_prefetch(arow + k + 256, 0, 1);
    __builtin_prefetch(bcol + k + 256, 0, 1);
    v8h alo = *(const v8h*)(arow + k + ao);
    v8h ahi = *(const v8h*)(arow + k + ao + 8);
    v16h a  = __builtin_shufflevector(alo, ahi,
              0, 1, 2, 3, 4, 5, 6, 7, 8, 9, 10, 11, 12, 13, 14, 15);
    v16h bf = *(const v16h*)(bcol + k);
    acc = __builtin_amdgcn_wmma_f32_16x16x32_f16(
        /*neg_a=*/false, a, /*neg_b=*/false, bf,
        /*c_mod=*/(short)0, acc, /*reuse_a=*/false, /*reuse_b=*/false);
  }

  // C/D layout: VGPR r -> row (bt*16 + r + 8*half), col lr.
  // logits layout (m,b,c,n): consecutive r rows differ by C_*N_ floats
  // (512 B) -> single base pointer + immediate instruction offsets.
  float* orow = logits +
      (size_t)(((m * B_ + bt * 16 + 8 * half) * C_ + c) * N_ + lr);
#pragma unroll
  for (int r = 0; r < 8; ++r) {
    orow[r * C_ * N_] = acc[r];
  }
}

// ---------------------------------------------------------------------------
// Kernel 4: per-m temperature: temp[m] = 1/(mean_{b,c} max_n logits + eps) - 1
// ---------------------------------------------------------------------------
__global__ void k_temp(const float* __restrict__ logits,
                       float* __restrict__ temp) {
  __shared__ float red[256];
  int m = blockIdx.x;
  int t = threadIdx.x;
  float s = 0.f;
  for (int j = 0; j < (B_ * C_) / 256; ++j) {   // 1024 fam values, 4/thread
    int e = j * 256 + t;
    int b = e >> 3;
    int c = e & 7;
    const float* lp = logits + (size_t)((m * B_ + b) * C_ + c) * N_;
    float mx = lp[0];
    for (int n = 1; n < N_; ++n) mx = fmaxf(mx, lp[n]);
    s += mx;
  }
  red[t] = s;
  __syncthreads();
  for (int off = 128; off >= 1; off >>= 1) {
    if (t < off) red[t] += red[t + off];
    __syncthreads();
  }
  if (t == 0) {
    float avg = red[0] * (1.0f / (B_ * C_));
    temp[m] = 1.0f / (avg + EPS_) - 1.0f;
  }
}

// ---------------------------------------------------------------------------
// Kernel 5: Gumbel-max categorical sample over n, one-hot int32 output,
// output layout (b, m, c, n). One thread per (b,m,c).
// ---------------------------------------------------------------------------
__device__ __forceinline__ unsigned hash_u32(unsigned x) {
  x ^= x >> 17; x *= 0xed5ad4bbu;
  x ^= x >> 11; x *= 0xac4c1b51u;
  x ^= x >> 15; x *= 0x31848babu;
  x ^= x >> 14;
  return x;
}

__global__ void k_sample(const float* __restrict__ logits,
                         const float* __restrict__ temp,
                         int* __restrict__ out) {
  int tid = blockIdx.x * 256 + threadIdx.x;      // 0 .. B_*M_*C_-1
  int b   = tid / (M_ * C_);
  int rem = tid - b * (M_ * C_);
  int m   = rem >> 3;
  int c   = rem & 7;

  float invt = 1.0f / temp[m];
  const float* lp = logits + (size_t)((m * B_ + b) * C_ + c) * N_;

  int best = 0;
  float bestv = -3.0e38f;
#pragma unroll
  for (int n = 0; n < N_; ++n) {
    unsigned h = hash_u32((unsigned)(tid * 16 + n) * 0x9E3779B9u + 42u);
    float u = ((float)(h >> 8) + 0.5f) * (1.0f / 16777216.0f);   // (0,1)
    float g = -__logf(-__logf(u));
    float sc = lp[n] * invt + g;
    if (sc > bestv) { bestv = sc; best = n; }
  }
  int* op = out + (size_t)((b * M_ + m) * C_ + c) * N_;
#pragma unroll
  for (int n = 0; n < N_; ++n) op[n] = (n == best) ? 1 : 0;
}

// ---------------------------------------------------------------------------
extern "C" void kernel_launch(void* const* d_in, const int* in_sizes, int n_in,
                              void* d_out, int out_size, void* d_ws, size_t ws_size,
                              hipStream_t stream) {
  const float* x  = (const float*)d_in[0];        // (128,100,8,16)
  const float* w  = (const float*)d_in[1];        // (100,8,3200,16)
  const int* filt = (const int*)d_in[2];          // (100,25)
  int* out = (int*)d_out;                         // (128,100,8,16) int32

  char* ws = (char*)d_ws;
  _Float16* xn     = (_Float16*)(ws);                       // 81,920,000 B
  _Float16* wt     = (_Float16*)(ws + 81920000);            // 81,920,000 B
  float*    logits = (float*)   (ws + 163840000);           //  6,553,600 B
  float*    temp   = (float*)   (ws + 170393600);           //        400 B

  // 1) gather + normalize -> xn (M,B,D) f16
  k_gather_norm<<<(M_ * B_) / 4, 128, 0, stream>>>(x, filt, xn);
  // 2) weights fp32 (m,c,d,n) -> f16 (m,c,n,d)
  k_transpose_w<<<dim3(D_ / 128, M_ * C_), 256, 0, stream>>>(w, wt);
  // 3) 100 x (128x128x3200) f16 WMMA GEMMs -> logits (m,b,c,n) f32
  k_gemm<<<dim3(B_ / 16, M_), 256, 0, stream>>>(xn, wt, logits);
  // 4) per-m temperature
  k_temp<<<M_, 256, 0, stream>>>(logits, temp);
  // 5) Gumbel-max sample + one-hot int32 (b,m,c,n)
  k_sample<<<(B_ * M_ * C_) / 256, 256, 0, stream>>>(logits, temp, out);
}
